// Decoder_7765300871422
// MI455X (gfx1250) — matrix operand order repaired and checked by reading, not compiled, not run on hardware
//
#include <hip/hip_runtime.h>
#include <hip/hip_bf16.h>

// ---------------------------------------------------------------------------
// CDNA5 (gfx1250) attention-LSTM greedy decoder.
// Heavy lifting: generic bf16 WMMA GEMM (v_wmma_f32_16x16x32_bf16, wave32,
// one 16x16 tile per wave, fp32 accumulate). Weights converted to bf16 once
// per launch; W_emb (32.8MB bf16) stays resident in the 192MB L2 across the
// 100 sequential decode steps.
// ---------------------------------------------------------------------------

typedef __attribute__((ext_vector_type(16))) __bf16 v16bf;
typedef __attribute__((ext_vector_type(8)))  __bf16 v8bf;
typedef __attribute__((ext_vector_type(8)))  float  v8f;

#define DEC_B   32
#define DEC_T   512
#define DEC_H   512
#define DEC_A   128
#define DEC_VD  512
#define DEC_V   32000
#define DEC_L   100
#define DEC_CD  1024          /* H + Vd */
#define DEC_XC  1536          /* H + Vd + H (concat input to [W_ih|W_hh]) */

__device__ __forceinline__ unsigned short f2bf(float f) {
    unsigned int u = __float_as_uint(f);
    u += 0x7FFFu + ((u >> 16) & 1u);          // round-to-nearest-even
    return (unsigned short)(u >> 16);
}

// ---------------------------------------------------------------------------
// Generic WMMA GEMM:  D[M,N] = A[M,K] * Bt[N,K]^T + bias   (all bf16 in,
// fp32 accum).  act==1 applies leaky_relu(0.01).  D (fp32, row stride ldd)
// and Dbf (bf16, row stride N) are each optional.
//
// Fragment layout assumptions (ISA 7.12.2, 16-bit A 16x32 / B 32x16):
//   A: lane = (m & 15) + 16*half ; elems j<8 -> K=k0+8*half+j,
//                                  j>=8 -> K=k0+16+8*half+(j-8)
//   B: lane = (n & 15) + 16*half ; same K mapping as A
// Both become two 16-byte global_load_b128 per fragment since Bt rows are
// contiguous in K.
// ---------------------------------------------------------------------------
__global__ __launch_bounds__(256)
void gemm_bf16_wmma(const unsigned short* __restrict__ A,
                    const unsigned short* __restrict__ Bt,
                    const float* __restrict__ bias,
                    float* __restrict__ D, long ldd,
                    unsigned short* __restrict__ Dbf,
                    int M, int N, int K, int act)
{
    const int lane = threadIdx.x & 31;
    const int wid  = threadIdx.x >> 5;
    const int Mt = M >> 4, Nt = N >> 4;
    int tile = blockIdx.x * 8 + wid;
    if (tile >= Mt * Nt) return;
    const int mt = tile % Mt;
    const int nt = tile / Mt;
    const int half = lane >> 4;
    const int l15  = lane & 15;

    const unsigned short* __restrict__ Arow = A  + (long)((mt << 4) + l15) * K;
    const unsigned short* __restrict__ Brow = Bt + (long)((nt << 4) + l15) * K;

    v8f acc = {};
    for (int k0 = 0; k0 < K; k0 += 32) {
        v8bf a_lo = *(const v8bf*)(Arow + k0 + 8 * half);
        v8bf a_hi = *(const v8bf*)(Arow + k0 + 16 + 8 * half);
        v8bf b_lo = *(const v8bf*)(Brow + k0 + 8 * half);
        v8bf b_hi = *(const v8bf*)(Brow + k0 + 16 + 8 * half);
        v16bf a, b;
#pragma unroll
        for (int j = 0; j < 8; ++j) {
            a[j] = a_lo[j]; a[j + 8] = a_hi[j];
            b[j] = b_lo[j]; b[j + 8] = b_hi[j];
        }
        acc = __builtin_amdgcn_wmma_f32_16x16x32_bf16(
                  false, a, false, b, (short)0, acc, false, false);
    }

    // C/D layout: VGPR e -> M = e + 8*half ; N = lane & 15
    const int ncol = (nt << 4) + l15;
    const float bv = bias ? bias[ncol] : 0.0f;
#pragma unroll
    for (int e = 0; e < 8; ++e) {
        int m = (mt << 4) + e + 8 * half;
        float v = acc[e] + bv;
        if (act == 1) v = (v > 0.0f) ? v : 0.01f * v;
        if (D)   D[(long)m * ldd + ncol] = v;
        if (Dbf) Dbf[(long)m * N + ncol] = f2bf(v);
    }
}

// ---------------------------------------------------------------------------
// Weight conversion / packing (once per launch)
// ---------------------------------------------------------------------------
__global__ void f32_to_bf16_kernel(const float* __restrict__ src,
                                   unsigned short* __restrict__ dst, long n)
{
    long i = (long)blockIdx.x * blockDim.x + threadIdx.x;
    long stride = (long)gridDim.x * blockDim.x;
    for (; i < n; i += stride) dst[i] = f2bf(src[i]);
}

// Wg[n, 0:1024] = W_ih[n,:], Wg[n, 1024:1536] = W_hh[n,:]   (bf16)
__global__ void concat_wg_kernel(const float* __restrict__ W_ih,
                                 const float* __restrict__ W_hh,
                                 unsigned short* __restrict__ Wg)
{
    long total = (long)(4 * DEC_H) * DEC_XC;
    long i = (long)blockIdx.x * blockDim.x + threadIdx.x;
    long stride = (long)gridDim.x * blockDim.x;
    for (; i < total; i += stride) {
        int n = (int)(i / DEC_XC), k = (int)(i % DEC_XC);
        float v = (k < DEC_CD) ? W_ih[(long)n * DEC_CD + k]
                               : W_hh[(long)n * DEC_H + (k - DEC_CD)];
        Wg[i] = f2bf(v);
    }
}

__global__ void init_state_kernel(const float* __restrict__ h00,
                                  const float* __restrict__ c00,
                                  unsigned short* __restrict__ h_bf,
                                  float* __restrict__ c_f32,
                                  int* __restrict__ idx)
{
    int i = blockIdx.x * blockDim.x + threadIdx.x;     // 32*512
    if (i >= DEC_B * DEC_H) return;
    int u = i & (DEC_H - 1);
    h_bf[i]  = f2bf(h00[u]);
    c_f32[i] = c00[u];
    if (i < DEC_B) idx[i] = 0;
}

// xcat[b] = [ emb(bf16 W_emb[idx[b]]) | ctx | h ]   (bf16, 1536)
__global__ void pack_xcat_kernel(const unsigned short* __restrict__ W_emb_bf,
                                 const int* __restrict__ idx,
                                 const unsigned short* __restrict__ ctx_bf,
                                 const unsigned short* __restrict__ h_bf,
                                 unsigned short* __restrict__ xcat)
{
    int i = blockIdx.x * blockDim.x + threadIdx.x;     // 32*1536
    if (i >= DEC_B * DEC_XC) return;
    int b = i / DEC_XC, k = i % DEC_XC;
    unsigned short v;
    if (k < DEC_H)        v = W_emb_bf[(long)idx[b] * DEC_H + k];
    else if (k < DEC_CD)  v = ctx_bf[b * DEC_VD + (k - DEC_H)];
    else                  v = h_bf[b * DEC_H + (k - DEC_CD)];
    xcat[i] = v;
}

// LSTM cell elementwise (i,f,g,o order).  Writes c (fp32), h (bf16) and
// the c1 half of cat2 = [c1 | ctx1] (bf16).
__global__ void lstm_cell_kernel(const float* __restrict__ gates,
                                 const float* __restrict__ b_ih,
                                 const float* __restrict__ b_hh,
                                 float* __restrict__ c_f32,
                                 unsigned short* __restrict__ h_bf,
                                 unsigned short* __restrict__ cat2_bf)
{
    int i = blockIdx.x * blockDim.x + threadIdx.x;     // 32*512
    if (i >= DEC_B * DEC_H) return;
    int b = i >> 9, u = i & (DEC_H - 1);
    const float* g = gates + (long)b * 4 * DEC_H;
    float ig = g[u]              + b_ih[u]              + b_hh[u];
    float fg = g[DEC_H + u]      + b_ih[DEC_H + u]      + b_hh[DEC_H + u];
    float gg = g[2 * DEC_H + u]  + b_ih[2 * DEC_H + u]  + b_hh[2 * DEC_H + u];
    float og = g[3 * DEC_H + u]  + b_ih[3 * DEC_H + u]  + b_hh[3 * DEC_H + u];
    float si = 1.0f / (1.0f + __expf(-ig));
    float sf = 1.0f / (1.0f + __expf(-fg));
    float so = 1.0f / (1.0f + __expf(-og));
    float c1 = sf * c_f32[i] + si * tanhf(gg);
    float h1 = so * tanhf(c1);
    c_f32[i] = c1;
    h_bf[i]  = f2bf(h1);
    cat2_bf[b * DEC_CD + u] = f2bf(c1);
}

// ---------------------------------------------------------------------------
// Attention: energy = q . key[:, :, t] per t, masked softmax (== reference's
// softmax*mask renormalized), ctx = att @ value.   One block per batch row.
// ---------------------------------------------------------------------------
__global__ __launch_bounds__(512)
void attend_kernel(const float* __restrict__ q,       // [32,128], bq included
                   const float* __restrict__ key,     // [B,A,T]
                   const float* __restrict__ value,   // [B,T,Vd]
                   const int* __restrict__ src_lens,
                   unsigned short* __restrict__ ctx_bf,   // [32,512]
                   unsigned short* __restrict__ cat2_bf)  // [32,1024], slot [512:)
{
    __shared__ float s_att[DEC_T];
    __shared__ float s_red[DEC_T];
    const int b = blockIdx.x;
    const int tid = threadIdx.x;
    const int len = src_lens[b];

    const float* kb = key + (long)b * DEC_A * DEC_T;
    const float* qb = q + b * DEC_A;
    float e = 0.0f;
    for (int a = 0; a < DEC_A; ++a)             // coalesced across tid
        e += qb[a] * kb[a * DEC_T + tid];
    const bool valid = tid < len;

    s_red[tid] = valid ? e : -3.4e38f;
    __syncthreads();
    for (int s = DEC_T / 2; s > 0; s >>= 1) {
        if (tid < s) s_red[tid] = fmaxf(s_red[tid], s_red[tid + s]);
        __syncthreads();
    }
    const float mx = s_red[0];
    __syncthreads();

    float ex = valid ? __expf(e - mx) : 0.0f;
    s_red[tid] = ex;
    __syncthreads();
    for (int s = DEC_T / 2; s > 0; s >>= 1) {
        if (tid < s) s_red[tid] += s_red[tid + s];
        __syncthreads();
    }
    s_att[tid] = ex * (1.0f / s_red[0]);
    __syncthreads();

    const float* vb = value + (long)b * DEC_T * DEC_VD;
    float c = 0.0f;
    for (int t = 0; t < DEC_T; ++t)             // coalesced across tid
        c += s_att[t] * vb[t * DEC_VD + tid];
    unsigned short cb = f2bf(c);
    ctx_bf[b * DEC_VD + tid] = cb;
    cat2_bf[b * DEC_CD + DEC_H + tid] = cb;
}

// ---------------------------------------------------------------------------
// In-place log_softmax over d_out row [b, t, :] + greedy argmax -> idx[b].
// argmax taken before the shift (invariant).  First-occurrence tie-break.
// ---------------------------------------------------------------------------
__global__ __launch_bounds__(256)
void logsoftmax_argmax_kernel(float* __restrict__ out, long row_stride, int t,
                              int* __restrict__ idx)
{
    __shared__ float s_val[256];
    __shared__ int   s_idx[256];
    const int b = blockIdx.x;
    const int tid = threadIdx.x;
    float* row = out + (long)b * row_stride + (long)t * DEC_V;

    float mx = -3.4e38f; int mi = DEC_V;
    for (int j = tid; j < DEC_V; j += 256) {
        float v = row[j];
        if (v > mx) { mx = v; mi = j; }
    }
    s_val[tid] = mx; s_idx[tid] = mi;
    __syncthreads();
    for (int s = 128; s > 0; s >>= 1) {
        if (tid < s) {
            float v2 = s_val[tid + s]; int i2 = s_idx[tid + s];
            if (v2 > s_val[tid] || (v2 == s_val[tid] && i2 < s_idx[tid])) {
                s_val[tid] = v2; s_idx[tid] = i2;
            }
        }
        __syncthreads();
    }
    const float rmax = s_val[0];
    const int   rarg = s_idx[0];
    __syncthreads();

    float sum = 0.0f;
    for (int j = tid; j < DEC_V; j += 256) sum += __expf(row[j] - rmax);
    s_val[tid] = sum;
    __syncthreads();
    for (int s = 128; s > 0; s >>= 1) {
        if (tid < s) s_val[tid] += s_val[tid + s];
        __syncthreads();
    }
    const float lse = rmax + __logf(s_val[0]);
    for (int j = tid; j < DEC_V; j += 256) row[j] -= lse;
    if (tid == 0) idx[b] = rarg;
}

// ---------------------------------------------------------------------------
extern "C" void kernel_launch(void* const* d_in, const int* in_sizes, int n_in,
                              void* d_out, int out_size, void* d_ws, size_t ws_size,
                              hipStream_t stream)
{
    const float* key      = (const float*)d_in[0];
    const float* value    = (const float*)d_in[1];
    /* Yinput d_in[2], max_len d_in[3] unused (greedy eval, L=100 fixed) */
    const int*   src_lens = (const int*)d_in[4];
    const float* W_emb    = (const float*)d_in[5];
    const float* b_proj   = (const float*)d_in[6];
    const float* Wq       = (const float*)d_in[7];
    const float* bq       = (const float*)d_in[8];
    const float* W_ih     = (const float*)d_in[9];
    const float* W_hh     = (const float*)d_in[10];
    const float* b_ih     = (const float*)d_in[11];
    const float* b_hh     = (const float*)d_in[12];
    const float* Wm       = (const float*)d_in[13];
    const float* bm       = (const float*)d_in[14];
    const float* h00      = (const float*)d_in[15];
    const float* c00      = (const float*)d_in[16];
    float* out = (float*)d_out;

    // ---- workspace carve-out (256B aligned) ----
    char* p = (char*)d_ws;
    auto carve = [&](size_t bytes) -> void* {
        void* r = (void*)p;
        p += (bytes + 255) & ~(size_t)255;
        return r;
    };
    unsigned short* W_emb_bf = (unsigned short*)carve((size_t)DEC_V * DEC_H * 2);
    unsigned short* Wg_bf    = (unsigned short*)carve((size_t)4 * DEC_H * DEC_XC * 2);
    unsigned short* Wq_bf    = (unsigned short*)carve((size_t)DEC_A * DEC_H * 2);
    unsigned short* Wm_bf    = (unsigned short*)carve((size_t)DEC_H * DEC_CD * 2);
    unsigned short* xcat_bf  = (unsigned short*)carve((size_t)DEC_B * DEC_XC * 2);
    unsigned short* cat2_bf  = (unsigned short*)carve((size_t)DEC_B * DEC_CD * 2);
    unsigned short* h_bf     = (unsigned short*)carve((size_t)DEC_B * DEC_H * 2);
    unsigned short* ctx_bf   = (unsigned short*)carve((size_t)DEC_B * DEC_VD * 2);
    unsigned short* proj_bf  = (unsigned short*)carve((size_t)DEC_B * DEC_H * 2);
    float* c_f32    = (float*)carve((size_t)DEC_B * DEC_H * 4);
    float* gates_f32= (float*)carve((size_t)DEC_B * 4 * DEC_H * 4);
    float* q_f32    = (float*)carve((size_t)DEC_B * DEC_A * 4);
    int*   idx      = (int*)carve((size_t)DEC_B * 4);

    // ---- one-time (per launch) weight conversion ----
    f32_to_bf16_kernel<<<2048, 256, 0, stream>>>(W_emb, W_emb_bf, (long)DEC_V * DEC_H);
    f32_to_bf16_kernel<<<256,  256, 0, stream>>>(Wq, Wq_bf, (long)DEC_A * DEC_H);
    f32_to_bf16_kernel<<<512,  256, 0, stream>>>(Wm, Wm_bf, (long)DEC_H * DEC_CD);
    concat_wg_kernel<<<2048, 256, 0, stream>>>(W_ih, W_hh, Wg_bf);
    init_state_kernel<<<64, 256, 0, stream>>>(h00, c00, h_bf, c_f32, idx);

    // ---- initial attention with h0 ----
    gemm_bf16_wmma<<<2, 256, 0, stream>>>(h_bf, Wq_bf, bq, q_f32, DEC_A,
                                          nullptr, DEC_B, DEC_A, DEC_H, 0);
    attend_kernel<<<DEC_B, DEC_T, 0, stream>>>(q_f32, key, value, src_lens,
                                               ctx_bf, cat2_bf);

    const long out_row_stride = (long)DEC_L * DEC_V;
    for (int t = 0; t < DEC_L; ++t) {
        // x = [emb|ctx|h]
        pack_xcat_kernel<<<192, 256, 0, stream>>>(W_emb_bf, idx, ctx_bf, h_bf, xcat_bf);
        // gates = xcat @ [W_ih|W_hh]^T          M=32 N=2048 K=1536 -> 256 tiles
        gemm_bf16_wmma<<<32, 256, 0, stream>>>(xcat_bf, Wg_bf, nullptr,
                                               gates_f32, 4 * DEC_H, nullptr,
                                               DEC_B, 4 * DEC_H, DEC_XC, 0);
        lstm_cell_kernel<<<64, 256, 0, stream>>>(gates_f32, b_ih, b_hh,
                                                 c_f32, h_bf, cat2_bf);
        // q = h1 @ Wq^T + bq                    M=32 N=128 K=512 -> 16 tiles
        gemm_bf16_wmma<<<2, 256, 0, stream>>>(h_bf, Wq_bf, bq, q_f32, DEC_A,
                                              nullptr, DEC_B, DEC_A, DEC_H, 0);
        attend_kernel<<<DEC_B, DEC_T, 0, stream>>>(q_f32, key, value, src_lens,
                                                   ctx_bf, cat2_bf);
        // proj = leaky_relu([c1|ctx1] @ Wm^T + bm)   M=32 N=512 K=1024 -> 64 tiles
        gemm_bf16_wmma<<<8, 256, 0, stream>>>(cat2_bf, Wm_bf, bm,
                                              nullptr, 0, proj_bf,
                                              DEC_B, DEC_H, DEC_CD, 1);
        // logits = proj @ W_emb^T + b_proj (into d_out)  M=32 N=32000 K=512 -> 4000 tiles
        gemm_bf16_wmma<<<500, 256, 0, stream>>>(proj_bf, W_emb_bf, b_proj,
                                                out + (long)t * DEC_V, out_row_stride,
                                                nullptr, DEC_B, DEC_V, DEC_H, 0);
        // log_softmax in place + greedy argmax -> next idx
        logsoftmax_argmax_kernel<<<DEC_B, 256, 0, stream>>>(out, out_row_stride, t, idx);
    }
}
